// multi_head_attention_71021579206938
// MI455X (gfx1250) — matrix operand order repaired and checked by reading, not compiled, not run on hardware
//
#include <hip/hip_runtime.h>
#include <stdint.h>

typedef __bf16 bf16_t;
typedef __attribute__((ext_vector_type(16))) __bf16 v16bf;
typedef __attribute__((ext_vector_type(8)))  __bf16 v8bf;
typedef __attribute__((ext_vector_type(8)))  float  v8f;

#define B_  4
#define E_  1024
#define L_  2048
#define H_  16
#define DH_ 64
// softmax in base-2 domain: scale = E^-0.5 * log2(e)
#define SCALE2_ 0.0450842206030264f
#define NEGBIG_ -1e30f

union V16  { v16bf v; v8bf h[2]; };
union U4H8 { uint4 u; bf16_t h[8]; };
union U2H4 { uint2 u; uint32_t w[2]; bf16_t h[4]; };

static __device__ __forceinline__ bf16_t f2bf(float x) { return (bf16_t)x; }

static __device__ __forceinline__ uint32_t pack2bf(float a, float b) {
#if __has_builtin(__builtin_amdgcn_cvt_pk_bf16_f32)
  typedef __attribute__((ext_vector_type(2))) __bf16 v2bf;
  v2bf t = __builtin_amdgcn_cvt_pk_bf16_f32(a, b);
  return __builtin_bit_cast(uint32_t, t);
#else
  union { bf16_t h[2]; uint32_t u; } x;
  x.h[0] = (bf16_t)a; x.h[1] = (bf16_t)b;
  return x.u;
#endif
}

static __device__ __forceinline__ v8f wmma_bf16(v16bf a, v16bf b, v8f c) {
  return __builtin_amdgcn_wmma_f32_16x16x32_bf16(false, a, false, b, (short)0, c,
                                                 false, false);
}

// ---------------- Tensor Data Mover (guarded) ----------------
#if __has_builtin(__builtin_amdgcn_tensor_load_to_lds)
#define HAVE_TDM 1
typedef __attribute__((ext_vector_type(4))) unsigned int u32x4;
typedef __attribute__((ext_vector_type(8))) int i32x8;
typedef __attribute__((ext_vector_type(4))) int i32x4;

// 2D tile load: dim0 = contiguous elements (2B each), dim1 = rows.
static __device__ __forceinline__ void tdm_load_2d(uint32_t lds_addr,
                                                   const void* gaddr,
                                                   uint32_t tile_d0,
                                                   uint32_t tile_d1,
                                                   uint32_t tensor_d0,
                                                   uint32_t tensor_d1,
                                                   uint32_t stride_d0) {
  uint64_t ga = (uint64_t)(uintptr_t)gaddr;
  u32x4 g0;
  g0[0] = 1u;                                            // count=1, user mode
  g0[1] = lds_addr;                                      // lds_addr (bytes)
  g0[2] = (uint32_t)(ga & 0xffffffffu);                  // global_addr[31:0]
  g0[3] = (uint32_t)((ga >> 32) & 0x01ffffffu) | (2u << 30);  // addr[56:32]|type=2
  i32x8 g1;
  g1[0] = (int)(1u << 16);                               // data_size=1 (2 bytes)
  g1[1] = (int)((tensor_d0 & 0xffffu) << 16);            // tensor_dim0[15:0]
  g1[2] = (int)(((tensor_d0 >> 16) & 0xffffu) | ((tensor_d1 & 0xffffu) << 16));
  g1[3] = (int)(((tensor_d1 >> 16) & 0xffffu) | ((tile_d0 & 0xffffu) << 16));
  g1[4] = (int)(tile_d1 & 0xffffu);                      // tile_dim1, tile_dim2=0
  g1[5] = (int)stride_d0;                                // tensor_dim0_stride lo
  g1[6] = 0;
  g1[7] = 0;
  i32x4 z4 = {0, 0, 0, 0};
#if defined(__clang_major__) && (__clang_major__ >= 23)
  i32x8 z8 = {0, 0, 0, 0, 0, 0, 0, 0};
  __builtin_amdgcn_tensor_load_to_lds(g0, g1, z4, z4, z8, 0);
#else
  __builtin_amdgcn_tensor_load_to_lds(g0, g1, z4, z4, 0);
#endif
}

static __device__ __forceinline__ void tdm_wait() {
#if __has_builtin(__builtin_amdgcn_s_wait_tensorcnt)
  __builtin_amdgcn_s_wait_tensorcnt(0);
#endif
}
#endif

// C[f,l] = sum_e W[e,f] * A[b,e,l]  (+bias[f]); A-frag = X (M=l), B-frag = W^T.
// Block: 256 thr (8 waves, 2(f) x 4(l)), tile 64(f) x 128(l), K-step 32.
// Double-buffered LDS: stage tile i+1 while WMMAs consume tile i.
template <bool AF32, bool OUTBF>
__global__ __launch_bounds__(256) void gemm_wt_kernel(
    const float* __restrict__ W, const void* __restrict__ Ain,
    void* __restrict__ Cout, const float* __restrict__ bias) {
  __shared__ bf16_t Ws[2][64][40];    // [buf][f][k]
  __shared__ bf16_t Xs[2][128][40];   // [buf][n][k]

  const int b    = blockIdx.z;
  const int m0   = blockIdx.x * 64;
  const int n0   = blockIdx.y * 128;
  const int tid  = threadIdx.x;
  const int lane = tid & 31;
  const int wv   = tid >> 5;
  const int wm   = (wv & 1) * 32;
  const int wn   = (wv >> 1) * 32;
  const int half = lane >> 4;
  const int lm   = lane & 15;

  const float*  Af = (const float*)Ain + (size_t)b * E_ * L_;
  const bf16_t* Ab = (const bf16_t*)Ain + (size_t)b * E_ * L_;

  auto stage = [&](int k0, int bi) {
#pragma unroll
    for (int i = 0; i < 2; ++i) {
      int p = tid + i * 256;
      int f = p & 63, kg = (p >> 6) * 4;
      U2H4 u;
      u.w[0] = pack2bf(W[(size_t)(k0 + kg + 0) * E_ + (m0 + f)],
                       W[(size_t)(k0 + kg + 1) * E_ + (m0 + f)]);
      u.w[1] = pack2bf(W[(size_t)(k0 + kg + 2) * E_ + (m0 + f)],
                       W[(size_t)(k0 + kg + 3) * E_ + (m0 + f)]);
      *(uint2*)&Ws[bi][f][kg] = u.u;
      if (k0 + 32 < E_)                     // prefetch tile i+2
        __builtin_prefetch(&W[(size_t)(k0 + 32 + kg) * E_ + (m0 + f)], 0, 1);
    }
#pragma unroll
    for (int i = 0; i < 4; ++i) {
      int p = tid + i * 256;
      int n = p & 127, kg = (p >> 7) * 4;
      U2H4 u;
      if (AF32) {
        u.w[0] = pack2bf(Af[(size_t)(k0 + kg + 0) * L_ + (n0 + n)],
                         Af[(size_t)(k0 + kg + 1) * L_ + (n0 + n)]);
        u.w[1] = pack2bf(Af[(size_t)(k0 + kg + 2) * L_ + (n0 + n)],
                         Af[(size_t)(k0 + kg + 3) * L_ + (n0 + n)]);
      } else {
#pragma unroll
        for (int j = 0; j < 4; ++j)
          u.h[j] = Ab[(size_t)(k0 + kg + j) * L_ + (n0 + n)];
      }
      *(uint2*)&Xs[bi][n][kg] = u.u;
      if (k0 + 32 < E_) {
        if (AF32)
          __builtin_prefetch(&Af[(size_t)(k0 + 32 + kg) * L_ + (n0 + n)], 0, 1);
        else
          __builtin_prefetch(&Ab[(size_t)(k0 + 32 + kg) * L_ + (n0 + n)], 0, 1);
      }
    }
  };

  v8f acc[2][2];
#pragma unroll
  for (int t = 0; t < 2; ++t)
#pragma unroll
    for (int s = 0; s < 2; ++s)
#pragma unroll
      for (int r = 0; r < 8; ++r) acc[t][s][r] = 0.0f;

  stage(0, 0);

  for (int k0 = 0, it = 0; k0 < E_; k0 += 32, ++it) {
    const int buf = it & 1;
    __syncthreads();                       // tile `buf` staged, prev reads done
    if (k0 + 32 < E_) stage(k0 + 32, buf ^ 1);

    V16 ax[2], bw[2];
#pragma unroll
    for (int t = 0; t < 2; ++t) {
      int n = wn + t * 16 + lm;
      ax[t].h[0] = *(const v8bf*)&Xs[buf][n][8 * half];
      ax[t].h[1] = *(const v8bf*)&Xs[buf][n][16 + 8 * half];
    }
#pragma unroll
    for (int s = 0; s < 2; ++s) {
      int f = wm + s * 16 + lm;
      bw[s].h[0] = *(const v8bf*)&Ws[buf][f][8 * half];
      bw[s].h[1] = *(const v8bf*)&Ws[buf][f][16 + 8 * half];
    }
#pragma unroll
    for (int t = 0; t < 2; ++t)
#pragma unroll
      for (int s = 0; s < 2; ++s)
        acc[t][s] = wmma_bf16(ax[t].v, bw[s].v, acc[t][s]);
  }

  float*  Cf = (float*)Cout + (size_t)b * E_ * L_;
  bf16_t* Cb = (bf16_t*)Cout + (size_t)b * E_ * L_;
#pragma unroll
  for (int t = 0; t < 2; ++t)
#pragma unroll
    for (int s = 0; s < 2; ++s) {
      int lb = n0 + wn + t * 16 + 8 * half;
      int f  = m0 + wm + s * 16 + lm;
      if (OUTBF) {
        uint4 u;
        u.x = pack2bf(acc[t][s][0], acc[t][s][1]);
        u.y = pack2bf(acc[t][s][2], acc[t][s][3]);
        u.z = pack2bf(acc[t][s][4], acc[t][s][5]);
        u.w = pack2bf(acc[t][s][6], acc[t][s][7]);
        *(uint4*)&Cb[(size_t)f * L_ + lb] = u;
      } else {
        float bv = bias ? bias[f] : 0.0f;
        float4 v0 = make_float4(acc[t][s][0] + bv, acc[t][s][1] + bv,
                                acc[t][s][2] + bv, acc[t][s][3] + bv);
        float4 v1 = make_float4(acc[t][s][4] + bv, acc[t][s][5] + bv,
                                acc[t][s][6] + bv, acc[t][s][7] + bv);
        *(float4*)&Cf[(size_t)f * L_ + lb]     = v0;
        *(float4*)&Cf[(size_t)f * L_ + lb + 4] = v1;
      }
    }
}

// Flash attention over kqv[b, h*64+d, l]: query = column j, keys i<=j.
// Double-buffered K/V tiles; TDM DMA of next block overlapped with compute.
__global__ __launch_bounds__(128) void attn_kernel(
    const bf16_t* __restrict__ kqv, bf16_t* __restrict__ tout) {
  __shared__ bf16_t Ktdk[2][64][64];   // [buf][d][key] : TDM-loadable tiles
  __shared__ bf16_t Ktkd[2][64][64];   // [buf][key][d]
  __shared__ bf16_t Qs[64][64];        // [q][d]
  __shared__ bf16_t Pb[4][16][80];

  const int qt   = blockIdx.x;
  const int h    = blockIdx.y;
  const int b    = blockIdx.z;
  const int tid  = threadIdx.x;
  const int lane = tid & 31;
  const int wv   = tid >> 5;
  const int half = lane >> 4;
  const int lm   = lane & 15;
  const int q0   = qt * 64 + wv * 16;
  const int nkb  = qt + 1;

  const bf16_t* base = kqv + (size_t)(b * E_ + h * DH_) * L_;   // [64][L]

  auto stageK = [&](int kb, int bi) {
#ifdef HAVE_TDM
    if (wv == 0)   // async DMA of [d][key] tile; waited at next iteration top
      tdm_load_2d((uint32_t)(uintptr_t)&Ktdk[bi][0][0], base + kb * 64,
                  /*tile_d0=*/64, /*tile_d1=*/64,
                  /*tensor_d0=*/L_, /*tensor_d1=*/64, /*stride_d0=*/L_);
#endif
#pragma unroll
    for (int i = 0; i < 4; ++i) {
      int p = tid + i * 128;
      int d = p >> 3, kv = (p & 7) * 8;
      U4H8 g;
      g.u = *(const uint4*)(base + (size_t)d * L_ + kb * 64 + kv);
#ifndef HAVE_TDM
      *(uint4*)&Ktdk[bi][d][kv] = g.u;
#endif
#pragma unroll
      for (int j = 0; j < 8; ++j) Ktkd[bi][kv + j][d] = g.h[j];
      if (kb + 1 < nkb)
        __builtin_prefetch(base + (size_t)d * L_ + (kb + 1) * 64 + kv, 0, 1);
    }
  };

  // stage Q tile (64q x 64d) -> Qs[q][d]
#pragma unroll
  for (int i = 0; i < 4; ++i) {
    int p = tid + i * 128;
    int d = p >> 3, qv = (p & 7) * 8;
    U4H8 g;
    g.u = *(const uint4*)(base + (size_t)d * L_ + qt * 64 + qv);
#pragma unroll
    for (int j = 0; j < 8; ++j) Qs[qv + j][d] = g.h[j];
  }
  stageK(0, 0);
  __syncthreads();

  V16 qf[2];
#pragma unroll
  for (int c = 0; c < 2; ++c) {
    qf[c].h[0] = *(const v8bf*)&Qs[wv * 16 + lm][c * 32 + 8 * half];
    qf[c].h[1] = *(const v8bf*)&Qs[wv * 16 + lm][c * 32 + 16 + 8 * half];
  }

  v8f o[4];
  float mrow[8], lrow[8];
#pragma unroll
  for (int dn = 0; dn < 4; ++dn)
#pragma unroll
    for (int r = 0; r < 8; ++r) o[dn][r] = 0.0f;
#pragma unroll
  for (int r = 0; r < 8; ++r) { mrow[r] = NEGBIG_; lrow[r] = 0.0f; }

  for (int kb = 0; kb < nkb; ++kb) {
    const int buf = kb & 1;
#ifdef HAVE_TDM
    if (wv == 0) tdm_wait();     // DMA for `buf` complete
#endif
    __syncthreads();             // tile `buf` visible; prev buf reads done
    if (kb + 1 < nkb) stageK(kb + 1, buf ^ 1);

    // ---- S = Q K^T
    v8f s[4];
#pragma unroll
    for (int kn = 0; kn < 4; ++kn) {
      V16 k0f, k1f;
      k0f.h[0] = *(const v8bf*)&Ktkd[buf][kn * 16 + lm][8 * half];
      k0f.h[1] = *(const v8bf*)&Ktkd[buf][kn * 16 + lm][16 + 8 * half];
      k1f.h[0] = *(const v8bf*)&Ktkd[buf][kn * 16 + lm][32 + 8 * half];
      k1f.h[1] = *(const v8bf*)&Ktkd[buf][kn * 16 + lm][32 + 16 + 8 * half];
      v8f z;
#pragma unroll
      for (int r = 0; r < 8; ++r) z[r] = 0.0f;
      z = wmma_bf16(qf[0].v, k0f.v, z);
      z = wmma_bf16(qf[1].v, k1f.v, z);
      s[kn] = z;
    }

    // ---- scale (base-2) + causal mask + row max
    float lmax[8];
#pragma unroll
    for (int r = 0; r < 8; ++r) lmax[r] = NEGBIG_;
#pragma unroll
    for (int kn = 0; kn < 4; ++kn)
#pragma unroll
      for (int r = 0; r < 8; ++r) {
        int key = kb * 64 + kn * 16 + lm;
        int j   = q0 + r + 8 * half;
        float v = (key <= j) ? s[kn][r] * SCALE2_ : NEGBIG_;
        s[kn][r] = v;
        lmax[r]  = fmaxf(lmax[r], v);
      }
#pragma unroll
    for (int r = 0; r < 8; ++r) {
      float v = lmax[r];
#pragma unroll
      for (int off = 8; off >= 1; off >>= 1) v = fmaxf(v, __shfl_xor(v, off, 32));
      float mn    = fmaxf(mrow[r], v);
      float alpha = exp2f(mrow[r] - mn);
      mrow[r] = mn;
      lrow[r] *= alpha;
#pragma unroll
      for (int dn = 0; dn < 4; ++dn) o[dn][r] *= alpha;
    }
    float lsum[8];
#pragma unroll
    for (int r = 0; r < 8; ++r) lsum[r] = 0.0f;
#pragma unroll
    for (int kn = 0; kn < 4; ++kn)
#pragma unroll
      for (int r = 0; r < 8; ++r) {
        float p = exp2f(s[kn][r] - mrow[r]);
        lsum[r] += p;
        Pb[wv][r + 8 * half][kn * 16 + lm] = f2bf(p);
      }
#pragma unroll
    for (int r = 0; r < 8; ++r) {
      float v = lsum[r];
#pragma unroll
      for (int off = 8; off >= 1; off >>= 1) v += __shfl_xor(v, off, 32);
      lrow[r] += v;
    }

    // ---- O += P V
#pragma unroll
    for (int kc = 0; kc < 2; ++kc) {
      V16 pf;
      pf.h[0] = *(const v8bf*)&Pb[wv][lm][kc * 32 + 8 * half];
      pf.h[1] = *(const v8bf*)&Pb[wv][lm][kc * 32 + 16 + 8 * half];
#pragma unroll
      for (int dn = 0; dn < 4; ++dn) {
        V16 vf;
        vf.h[0] = *(const v8bf*)&Ktdk[buf][dn * 16 + lm][kc * 32 + 8 * half];
        vf.h[1] = *(const v8bf*)&Ktdk[buf][dn * 16 + lm][kc * 32 + 16 + 8 * half];
        o[dn] = wmma_bf16(pf.v, vf.v, o[dn]);
      }
    }
  }

  // ---- normalize + vectorized bf16 store
  float rinv[8];
#pragma unroll
  for (int r = 0; r < 8; ++r) rinv[r] = 1.0f / lrow[r];
#pragma unroll
  for (int dn = 0; dn < 4; ++dn) {
    int d  = dn * 16 + lm;
    int jb = q0 + 8 * half;
    uint4 u;
    u.x = pack2bf(o[dn][0] * rinv[0], o[dn][1] * rinv[1]);
    u.y = pack2bf(o[dn][2] * rinv[2], o[dn][3] * rinv[3]);
    u.z = pack2bf(o[dn][4] * rinv[4], o[dn][5] * rinv[5]);
    u.w = pack2bf(o[dn][6] * rinv[6], o[dn][7] * rinv[7]);
    *(uint4*)(tout + (size_t)(b * E_ + h * DH_ + d) * L_ + jb) = u;
  }
}

extern "C" void kernel_launch(void* const* d_in, const int* in_sizes, int n_in,
                              void* d_out, int out_size, void* d_ws,
                              size_t ws_size, hipStream_t stream) {
  const float* x     = (const float*)d_in[0];
  const float* Wqkv  = (const float*)d_in[1];
  const float* Wproj = (const float*)d_in[2];
  const float* bproj = (const float*)d_in[3];

  bf16_t* kqvb = (bf16_t*)d_ws;                       // 16 MB
  bf16_t* tb   = kqvb + (size_t)B_ * E_ * L_;         // 16 MB
  float*  out  = (float*)d_out;

  dim3 gg(E_ / 64, L_ / 128, B_);
  gemm_wt_kernel<true, true><<<gg, 256, 0, stream>>>(Wqkv, x, kqvb, nullptr);

  dim3 ga(L_ / 64, H_, B_);
  attn_kernel<<<ga, 128, 0, stream>>>(kqvb, tb);

  gemm_wt_kernel<false, false><<<gg, 256, 0, stream>>>(Wproj, tb, out, bproj);
}
